// HyperGCN_17111149707511
// MI455X (gfx1250) — matrix operand order repaired and checked by reading, not compiled, not run on hardware
//
#include <hip/hip_runtime.h>
#include <stdint.h>
#include <math.h>

// ---------------- problem constants (from reference) ----------------
#define NT   20000          // utterances per modality
#define NN   60000          // nodes
#define ND   1024           // input dim
#define NH   512            // hidden dim
#define LAT  64             // latent dim
#define NEH  4000           // hyperedges
#define NNZ  120000         // hypergraph incidences
#define NE   160000         // gcn edges

typedef __attribute__((ext_vector_type(16))) __bf16 bf16x16;
typedef __attribute__((ext_vector_type(8)))  __bf16 bf16x8;
typedef __attribute__((ext_vector_type(8)))  float  f32x8;
typedef __attribute__((ext_vector_type(4)))  unsigned int u32x4;
typedef __attribute__((ext_vector_type(4)))  int i32x4;

// ---------------- gfx1250 async-to-LDS path (guarded) ----------------
#if defined(__gfx1250__) && __has_builtin(__builtin_amdgcn_global_load_async_to_lds_b128)
#define USE_ASYNC_LDS 1
typedef __attribute__((address_space(1))) i32x4 as1_i32x4;
typedef __attribute__((address_space(3))) i32x4 as3_i32x4;
#else
#define USE_ASYNC_LDS 0
#endif

__device__ __forceinline__ void cp16_g2l(void* lds, const void* g) {
#if USE_ASYNC_LDS
  __builtin_amdgcn_global_load_async_to_lds_b128((as1_i32x4*)g, (as3_i32x4*)lds,
                                                 0, 0);
#else
  *(u32x4*)lds = *(const u32x4*)g;
#endif
}

__device__ __forceinline__ void async_wait_all() {
#if USE_ASYNC_LDS
#if __has_builtin(__builtin_amdgcn_s_wait_asynccnt)
  __builtin_amdgcn_s_wait_asynccnt(0);
#else
  asm volatile("s_wait_asynccnt 0" ::: "memory");
#endif
#endif
}

// ---------------- WMMA helpers ----------------
__device__ __forceinline__ f32x8 wmma_bf16(bf16x16 a, bf16x16 b, f32x8 c) {
  // D = A(16x32) * B(32x16) + C(16x16), f32 accumulate
  return __builtin_amdgcn_wmma_f32_16x16x32_bf16(false, a, false, b, (short)0, c,
                                                 false, false);
}

// A fragment (16x32 bf16), ISA 7.12.2: lane m = lane&15; lanes<16 hold K 0-7/16-23,
// lanes>=16 hold K 8-15/24-31, contiguous pairs -> two 16B loads.
__device__ __forceinline__ bf16x16 ld_afrag(const __bf16* base, int stride, int k0,
                                            int lane) {
  const __bf16* p = base + (lane & 15) * stride + k0 + ((lane >> 4) << 3);
  bf16x8 lo = *(const bf16x8*)p;
  bf16x8 hi = *(const bf16x8*)(p + 16);
  return __builtin_shufflevector(lo, hi, 0, 1, 2, 3, 4, 5, 6, 7, 8, 9, 10, 11, 12,
                                 13, 14, 15);
}

// B fragment (32x16 bf16): column n = lane&15, same K order as A (K 0-7/16-23 for
// lanes<16, 8-15/24-31 for lanes>=16) -> two 16B loads from weight row W[n][k] (out = A @ W^T).
__device__ __forceinline__ bf16x16 ld_bfrag(const __bf16* base, int stride, int k0,
                                            int lane) {
  const __bf16* p = base + (lane & 15) * stride + k0 + ((lane >> 4) << 3);
  bf16x8 lo = *(const bf16x8*)p;
  bf16x8 hi = *(const bf16x8*)(p + 16);
  return __builtin_shufflevector(lo, hi, 0, 1, 2, 3, 4, 5, 6, 7, 8, 9, 10, 11, 12,
                                 13, 14, 15);
}

// ---------------- kernel 1: weights -> bf16 (spline_w fused with scaler) --------
__global__ void k_prep_weights(const float* cw, const float* bw, const float* sw,
                               const float* ss, const float* ew, const float* hw,
                               const float* gw, __bf16* cwb, __bf16* bwb,
                               __bf16* swb, __bf16* ewb, __bf16* hwb, __bf16* gwb) {
  const long S0 = LAT * ND;            // compress_w
  const long S1 = LAT * LAT;           // base_w
  const long S2 = LAT * (LAT * 8);     // spline_w * scaler, reshaped [64][512]
  const long S3 = NH * LAT;            // expand_w
  const long S4 = 3L * NH * NH;        // hw
  const long S5 = 4L * NH * NH;        // gw
  const long TOT = S0 + S1 + S2 + S3 + S4 + S5;
  long stride = (long)gridDim.x * blockDim.x;
  for (long i = (long)blockIdx.x * blockDim.x + threadIdx.x; i < TOT; i += stride) {
    long j = i;
    if (j < S0) { cwb[j] = (__bf16)cw[j]; continue; } j -= S0;
    if (j < S1) { bwb[j] = (__bf16)bw[j]; continue; } j -= S1;
    if (j < S2) {
      long n = j / (LAT * 8), r = j % (LAT * 8), in = r / 8;
      swb[j] = (__bf16)(sw[j] * ss[n * LAT + in]);
      continue;
    } j -= S2;
    if (j < S3) { ewb[j] = (__bf16)ew[j]; continue; } j -= S3;
    if (j < S4) { hwb[j] = (__bf16)hw[j]; continue; } j -= S4;
    gwb[j] = (__bf16)gw[j];
  }
}

// ---------------- kernel 2: build X = concat(a, v, l + spk_emb) as bf16 --------
__global__ void k_build_x(const float* a, const float* v, const float* l,
                          const int* spk, const float* semb, __bf16* xb) {
  const long TOT = (long)NN * ND;
  long stride = (long)gridDim.x * blockDim.x;
  for (long i = (long)blockIdx.x * blockDim.x + threadIdx.x; i < TOT; i += stride) {
    int n = (int)(i >> 10), c = (int)(i & 1023);
    float val;
    if (n < NT) val = a[i];
    else if (n < 2 * NT) val = v[i - (long)NT * ND];
    else {
      int r = n - 2 * NT;
      val = l[(long)r * ND + c] + semb[(long)spk[r] * ND + c];
    }
    xb[i] = (__bf16)val;
  }
}

// ---------------- degree helpers ----------------
__global__ void k_zero(float* p, long n) {
  long stride = (long)gridDim.x * blockDim.x;
  for (long i = (long)blockIdx.x * blockDim.x + threadIdx.x; i < n; i += stride)
    p[i] = 0.0f;
}
__global__ void k_count(const int* idx, int nnz, float* deg) {
  long stride = (long)gridDim.x * blockDim.x;
  for (long i = (long)blockIdx.x * blockDim.x + threadIdx.x; i < nnz; i += stride)
    atomicAdd(&deg[idx[i]], 1.0f);
}
__global__ void k_invdeg(float* d, int n) {
  long stride = (long)gridDim.x * blockDim.x;
  for (long i = (long)blockIdx.x * blockDim.x + threadIdx.x; i < n; i += stride)
    d[i] = d[i] > 0.0f ? 1.0f / d[i] : 0.0f;
}
__global__ void k_rsqdeg(float* d, int n) {
  long stride = (long)gridDim.x * blockDim.x;
  for (long i = (long)blockIdx.x * blockDim.x + threadIdx.x; i < n; i += stride)
    d[i] = rsqrtf(d[i] + 1.0f);
}

// --------- kernel 3: compress GEMM (K=1024) + LN1 + KAN-LN + GELU + B-splines ---
// block = 128 threads (4 waves), 16 rows per block, 3750 blocks.
__global__ __launch_bounds__(128) void k_compress(
    const __bf16* __restrict__ xb, const __bf16* __restrict__ cwb,
    const float* __restrict__ cb, const float* __restrict__ g1,
    const float* __restrict__ b1, const float* __restrict__ g2,
    const float* __restrict__ b2, __bf16* __restrict__ xg,
    __bf16* __restrict__ sb) {
  __shared__ __align__(16) __bf16 At[16][ND];   // 32 KB
  __shared__ __align__(16) float Ct[16][LAT];   // 4 KB
  const int tid = threadIdx.x, lane = tid & 31, wave = tid >> 5;
  const int row0 = blockIdx.x * 16;

  // stage 16x1024 bf16 A tile (async-to-LDS when available)
  for (int p = 0; p < 16; ++p) {
    int idx = p * 128 + tid;              // 0..2047 16B segments
    int r = idx >> 7, seg = idx & 127;    // 128 segs per row
    cp16_g2l(&At[r][seg * 8], xb + (long)(row0 + r) * ND + seg * 8);
  }
  async_wait_all();
  __syncthreads();

  // each wave computes one 16x16 N-tile over K=1024 (32 WMMA steps)
  const int n0 = wave * 16;
  f32x8 acc = {};
  for (int kt = 0; kt < ND; kt += 32) {
    bf16x16 af = ld_afrag(&At[0][0], ND, kt, lane);
    bf16x16 bf = ld_bfrag(cwb + (long)n0 * ND, ND, kt, lane);
    acc = wmma_bf16(af, bf, acc);
  }
  {
    int nl = lane & 15, mh = (lane >> 4) << 3;
    float bias = cb[n0 + nl];
    for (int r = 0; r < 8; ++r) Ct[mh + r][n0 + nl] = acc[r] + bias;
  }
  __syncthreads();

  // LayerNorm x2 + GELU + spline bases: 8 lanes per row, 8 columns per lane
  {
    int r = tid >> 3;             // row 0..15
    int cg = (tid & 7) * 8;       // column base
    float x[8];
    for (int q = 0; q < 8; ++q) x[q] = Ct[r][cg + q];
    for (int pass = 0; pass < 2; ++pass) {
      float s = 0.f;
      for (int q = 0; q < 8; ++q) s += x[q];
      s += __shfl_xor(s, 1, 8); s += __shfl_xor(s, 2, 8); s += __shfl_xor(s, 4, 8);
      float mean = s * (1.0f / 64.0f);
      float vv = 0.f;
      for (int q = 0; q < 8; ++q) { float d = x[q] - mean; vv += d * d; }
      vv += __shfl_xor(vv, 1, 8); vv += __shfl_xor(vv, 2, 8); vv += __shfl_xor(vv, 4, 8);
      float inv = rsqrtf(vv * (1.0f / 64.0f) + 1e-5f);
      const float* gg = pass ? g2 : g1;
      const float* bb = pass ? b2 : b1;
      for (int q = 0; q < 8; ++q)
        x[q] = (x[q] - mean) * inv * gg[cg + q] + bb[cg + q];
    }
    // GELU (exact erf) -> base-branch input, bf16
    __bf16 ge[8];
    for (int q = 0; q < 8; ++q) {
      float t = x[q];
      ge[q] = (__bf16)(0.5f * t * (1.0f + erff(t * 0.70710678118f)));
    }
    *(bf16x8*)&xg[(long)(row0 + r) * LAT + cg] = *(const bf16x8*)ge;
    // Cox-de Boor cubic B-spline bases (grid 5, k=3 -> 8 bases per feature)
    for (int q = 0; q < 8; ++q) {
      float xv = x[q];
      float tt[12];
      for (int i = 0; i < 12; ++i) tt[i] = -2.2f + 0.4f * (float)i;
      float bas[11];
      for (int i = 0; i < 11; ++i)
        bas[i] = (xv >= tt[i] && xv < tt[i + 1]) ? 1.0f : 0.0f;
      for (int j = 1; j <= 3; ++j)
        for (int i = 0; i + j < 11; ++i)
          bas[i] = (xv - tt[i]) / (tt[i + j] - tt[i]) * bas[i] +
                   (tt[i + j + 1] - xv) / (tt[i + j + 1] - tt[i + 1]) * bas[i + 1];
      __bf16 outb[8];
      for (int i = 0; i < 8; ++i) outb[i] = (__bf16)bas[i];
      *(bf16x8*)&sb[(long)(row0 + r) * (LAT * 8) + (long)(cg + q) * 8] =
          *(const bf16x8*)outb;
    }
  }
}

// --------- kernel 4: mid = gelu(xk)@base_w^T + bases@sw^T ; H0 = mid@expand_w^T+b
__global__ __launch_bounds__(128) void k_kan_expand(
    const __bf16* __restrict__ xg, const __bf16* __restrict__ sb,
    const __bf16* __restrict__ bwb, const __bf16* __restrict__ swb,
    const __bf16* __restrict__ ewb, const float* __restrict__ eb,
    __bf16* __restrict__ h0) {
  __shared__ __align__(16) __bf16 XGt[16][LAT];       // 2 KB
  __shared__ __align__(16) __bf16 SBt[16][LAT * 8];   // 16 KB
  __shared__ __align__(16) __bf16 Mid[16][LAT];       // 2 KB
  const int tid = threadIdx.x, lane = tid & 31, wave = tid >> 5;
  const int row0 = blockIdx.x * 16;

  cp16_g2l(&XGt[tid >> 3][(tid & 7) * 8],
           xg + (long)(row0 + (tid >> 3)) * LAT + (tid & 7) * 8);
  for (int p = 0; p < 8; ++p) {
    int idx = p * 128 + tid;            // 1024 segs
    int r = idx >> 6, seg = idx & 63;
    cp16_g2l(&SBt[r][seg * 8], sb + (long)(row0 + r) * (LAT * 8) + seg * 8);
  }
  async_wait_all();
  __syncthreads();

  // phase 1: each wave -> one 16x16 tile of mid (K=64 base + K=512 spline)
  const int n0 = wave * 16;
  f32x8 acc = {};
  for (int kt = 0; kt < LAT; kt += 32)
    acc = wmma_bf16(ld_afrag(&XGt[0][0], LAT, kt, lane),
                    ld_bfrag(bwb + (long)n0 * LAT, LAT, kt, lane), acc);
  for (int kt = 0; kt < LAT * 8; kt += 32)
    acc = wmma_bf16(ld_afrag(&SBt[0][0], LAT * 8, kt, lane),
                    ld_bfrag(swb + (long)n0 * (LAT * 8), LAT * 8, kt, lane), acc);
  {
    int nl = lane & 15, mh = (lane >> 4) << 3;
    for (int r = 0; r < 8; ++r) Mid[mh + r][n0 + nl] = (__bf16)acc[r];
  }
  __syncthreads();

  // phase 2: expand to 512 (each wave: 8 N-tiles, K=64)
  for (int j = 0; j < 8; ++j) {
    int nn0 = (wave * 8 + j) * 16;
    f32x8 a2 = {};
    for (int kt = 0; kt < LAT; kt += 32)
      a2 = wmma_bf16(ld_afrag(&Mid[0][0], LAT, kt, lane),
                     ld_bfrag(ewb + (long)nn0 * LAT, LAT, kt, lane), a2);
    int nl = lane & 15, mh = (lane >> 4) << 3;
    float bias = eb[nn0 + nl];
    for (int r = 0; r < 8; ++r)
      h0[(long)(row0 + mh + r) * NH + nn0 + nl] = (__bf16)(a2[r] + bias);
  }
}

// --------- kernel 5: big GEMM out[M,512] = A[M,512] @ W[512,512]^T (bf16->bf16) -
// block 256 thr = 8 waves (4M x 2N), tile 64x128, k-step 32, LDS-staged.
__global__ __launch_bounds__(256) void k_gemm512(const __bf16* __restrict__ A,
                                                 const __bf16* __restrict__ W,
                                                 __bf16* __restrict__ out, int M) {
  __shared__ __align__(16) __bf16 At[64][32];    // 4 KB
  __shared__ __align__(16) __bf16 Bt[128][32];   // 8 KB
  const int tid = threadIdx.x, lane = tid & 31, wave = tid >> 5;
  const int wm = wave & 3, wn = wave >> 2;
  const int row0 = blockIdx.y * 64, col0 = blockIdx.x * 128;
  f32x8 acc0 = {}, acc1 = {}, acc2 = {}, acc3 = {};
  for (int kt = 0; kt < NH; kt += 32) {
    __syncthreads();
    {
      int r = tid >> 2, seg = tid & 3;                 // 64 rows x 4 segs
      int gr = row0 + r; if (gr >= M) gr = M - 1;
      cp16_g2l(&At[r][seg * 8], A + (long)gr * NH + kt + seg * 8);
    }
    for (int p = 0; p < 2; ++p) {                      // 128 rows x 4 segs
      int idx = p * 256 + tid;
      int r = idx >> 2, seg = idx & 3;
      cp16_g2l(&Bt[r][seg * 8], W + (long)(col0 + r) * NH + kt + seg * 8);
    }
    async_wait_all();
    __syncthreads();
    if (kt + 32 < NH) {   // hint next k-slice toward L2/L0
      int gr = row0 + (tid >> 2); if (gr >= M) gr = M - 1;
      __builtin_prefetch(A + (long)gr * NH + kt + 32, 0, 1);
      __builtin_prefetch(W + (long)(col0 + (tid >> 1)) * NH + kt + 32, 0, 1);
    }
    bf16x16 af = ld_afrag(&At[wm * 16][0], 32, 0, lane);
    acc0 = wmma_bf16(af, ld_bfrag(&Bt[wn * 64 + 0][0], 32, 0, lane), acc0);
    acc1 = wmma_bf16(af, ld_bfrag(&Bt[wn * 64 + 16][0], 32, 0, lane), acc1);
    acc2 = wmma_bf16(af, ld_bfrag(&Bt[wn * 64 + 32][0], 32, 0, lane), acc2);
    acc3 = wmma_bf16(af, ld_bfrag(&Bt[wn * 64 + 48][0], 32, 0, lane), acc3);
  }
  int nl = lane & 15, mh = (lane >> 4) << 3;
  f32x8 accs[4] = {acc0, acc1, acc2, acc3};
  for (int j = 0; j < 4; ++j)
    for (int r = 0; r < 8; ++r) {
      int grow = row0 + wm * 16 + mh + r;
      if (grow < M)
        out[(long)grow * NH + col0 + wn * 64 + j * 16 + nl] = (__bf16)accs[j][r];
    }
}

// ---------------- graph-phase kernels (bandwidth-bound, f32 atomics) ------------
__global__ void k_hyper_scatter1(const int* __restrict__ hn,
                                 const int* __restrict__ he,
                                 const __bf16* __restrict__ xt,
                                 float* __restrict__ m) {
  const long TOT = (long)NNZ * 128;
  long stride = (long)gridDim.x * blockDim.x;
  for (long i = (long)blockIdx.x * blockDim.x + threadIdx.x; i < TOT; i += stride) {
    int e = (int)(i >> 7), c4 = (int)(i & 127) << 2;
    const __bf16* src = xt + (long)hn[e] * NH + c4;
    float* dst = m + (long)he[e] * NH + c4;
    for (int j = 0; j < 4; ++j) atomicAdd(dst + j, (float)src[j]);
  }
}
__global__ void k_hyper_scatter2(const int* __restrict__ hn,
                                 const int* __restrict__ he,
                                 const float* __restrict__ m,
                                 const float* __restrict__ binv,
                                 float* __restrict__ acc) {
  const long TOT = (long)NNZ * 128;
  long stride = (long)gridDim.x * blockDim.x;
  for (long i = (long)blockIdx.x * blockDim.x + threadIdx.x; i < TOT; i += stride) {
    int e = (int)(i >> 7), c4 = (int)(i & 127) << 2;
    float w = binv[he[e]];
    const float* src = m + (long)he[e] * NH + c4;
    float* dst = acc + (long)hn[e] * NH + c4;
    for (int j = 0; j < 4; ++j) atomicAdd(dst + j, w * src[j]);
  }
}
__global__ void k_hyper_finish(const float* __restrict__ acc,
                               const float* __restrict__ dinv,
                               const float* __restrict__ hb,
                               __bf16* __restrict__ h0) {
  const long TOT = (long)NN * NH;
  long stride = (long)gridDim.x * blockDim.x;
  for (long i = (long)blockIdx.x * blockDim.x + threadIdx.x; i < TOT; i += stride) {
    int n = (int)(i >> 9);
    float v = dinv[n] * acc[i] + hb[i & (NH - 1)];
    h0[i] = (__bf16)fmaxf(v, 0.0f);
  }
}
__global__ void k_gcn_scatter(const int* __restrict__ es, const int* __restrict__ ed,
                              const __bf16* __restrict__ xt,
                              const float* __restrict__ dinv,
                              float* __restrict__ acc) {
  const long TOT = (long)NE * 128;
  long stride = (long)gridDim.x * blockDim.x;
  for (long i = (long)blockIdx.x * blockDim.x + threadIdx.x; i < TOT; i += stride) {
    int e = (int)(i >> 7), c4 = (int)(i & 127) << 2;
    int s = es[e], d = ed[e];
    float w = dinv[s] * dinv[d];
    const __bf16* src = xt + (long)s * NH + c4;
    float* dst = acc + (long)d * NH + c4;
    for (int j = 0; j < 4; ++j) atomicAdd(dst + j, w * (float)src[j]);
  }
}
__global__ void k_gcn_finish(const float* __restrict__ acc,
                             const __bf16* __restrict__ xt,
                             const float* __restrict__ dinv,
                             __bf16* __restrict__ h0, float* __restrict__ outf) {
  const long TOT = (long)NN * NH;
  long stride = (long)gridDim.x * blockDim.x;
  for (long i = (long)blockIdx.x * blockDim.x + threadIdx.x; i < TOT; i += stride) {
    int n = (int)(i >> 9);
    float dd = dinv[n];
    float v = fmaxf(acc[i] + (float)xt[i] * dd * dd, 0.0f);
    if (outf) outf[i] = v; else h0[i] = (__bf16)v;
  }
}

// ---------------- host launcher ----------------
static inline int nblk(long tot, int thr) { return (int)((tot + thr - 1) / thr); }

extern "C" void kernel_launch(void* const* d_in, const int* in_sizes, int n_in,
                              void* d_out, int out_size, void* d_ws, size_t ws_size,
                              hipStream_t stream) {
  const float* a    = (const float*)d_in[0];
  const float* v    = (const float*)d_in[1];
  const float* l    = (const float*)d_in[2];
  const int*   spk  = (const int*)d_in[3];
  const float* semb = (const float*)d_in[4];
  const float* cw   = (const float*)d_in[5];
  const float* cb   = (const float*)d_in[6];
  const float* g1   = (const float*)d_in[7];
  const float* b1   = (const float*)d_in[8];
  const float* g2   = (const float*)d_in[9];
  const float* b2   = (const float*)d_in[10];
  const float* bw   = (const float*)d_in[11];
  const float* sw   = (const float*)d_in[12];
  const float* ss   = (const float*)d_in[13];
  const float* ew   = (const float*)d_in[14];
  const float* eb   = (const float*)d_in[15];
  const float* hw   = (const float*)d_in[16];
  const float* hb   = (const float*)d_in[17];
  const float* gw   = (const float*)d_in[18];
  const int* hn     = (const int*)d_in[19];
  const int* he     = (const int*)d_in[20];
  const int* es     = (const int*)d_in[21];
  const int* ed     = (const int*)d_in[22];
  float* outf       = (float*)d_out;
  (void)in_sizes; (void)n_in; (void)out_size; (void)ws_size;

  // bump allocator over d_ws
  char* ws = (char*)d_ws;
  size_t off = 0;
  auto alloc = [&](size_t bytes) -> void* {
    void* p = ws + off;
    off = (off + bytes + 255) & ~(size_t)255;
    return p;
  };
  __bf16* XB   = (__bf16*)alloc((size_t)NN * ND * 2);      // 122.88 MB (dead after compress)
  float*  ACC  = (float*)XB;                               // alias: NN*NH*4 == NN*ND*2
  __bf16* CWB  = (__bf16*)alloc((size_t)LAT * ND * 2);
  __bf16* BWB  = (__bf16*)alloc((size_t)LAT * LAT * 2);
  __bf16* SWB  = (__bf16*)alloc((size_t)LAT * LAT * 8 * 2);
  __bf16* EWB  = (__bf16*)alloc((size_t)NH * LAT * 2);
  __bf16* HWB  = (__bf16*)alloc((size_t)3 * NH * NH * 2);
  __bf16* GWB  = (__bf16*)alloc((size_t)4 * NH * NH * 2);
  __bf16* XG   = (__bf16*)alloc((size_t)NN * LAT * 2);
  __bf16* SB   = (__bf16*)alloc((size_t)NN * LAT * 8 * 2);
  __bf16* H0   = (__bf16*)alloc((size_t)NN * NH * 2);      // activations (ping)
  __bf16* H1   = (__bf16*)alloc((size_t)NN * NH * 2);      // xt (pong)
  float*  MBUF = (float*)alloc((size_t)NEH * NH * 4);
  float*  DINV = (float*)alloc((size_t)NN * 4);
  float*  BINV = (float*)alloc((size_t)NEH * 4);
  float*  GDNV = (float*)alloc((size_t)NN * 4);

  // ---- prep: weights to bf16, degree tables ----
  k_prep_weights<<<4096, 256, 0, stream>>>(cw, bw, sw, ss, ew, hw, gw, CWB, BWB,
                                           SWB, EWB, HWB, GWB);
  k_build_x<<<nblk((long)NN * ND, 256), 256, 0, stream>>>(a, v, l, spk, semb, XB);
  k_zero<<<nblk(NN, 256), 256, 0, stream>>>(DINV, NN);
  k_zero<<<nblk(NEH, 256), 256, 0, stream>>>(BINV, NEH);
  k_zero<<<nblk(NN, 256), 256, 0, stream>>>(GDNV, NN);
  k_count<<<nblk(NNZ, 256), 256, 0, stream>>>(hn, NNZ, DINV);
  k_count<<<nblk(NNZ, 256), 256, 0, stream>>>(he, NNZ, BINV);
  k_count<<<nblk(NE, 256), 256, 0, stream>>>(ed, NE, GDNV);
  k_invdeg<<<nblk(NN, 256), 256, 0, stream>>>(DINV, NN);
  k_invdeg<<<nblk(NEH, 256), 256, 0, stream>>>(BINV, NEH);
  k_rsqdeg<<<nblk(NN, 256), 256, 0, stream>>>(GDNV, NN);

  // ---- bottleneck KAN (fc1) ----
  k_compress<<<NN / 16, 128, 0, stream>>>(XB, CWB, cb, g1, b1, g2, b2, XG, SB);
  k_kan_expand<<<NN / 16, 128, 0, stream>>>(XG, SB, BWB, SWB, EWB, eb, H0);

  dim3 ggrid(4, (NN + 63) / 64);   // 4 x 938 tiles of 64x128
  const long featTot = (long)NN * NH;

  // ---- 3 hypergraph conv layers ----
  for (int i = 0; i < 3; ++i) {
    k_gemm512<<<ggrid, 256, 0, stream>>>(H0, HWB + (size_t)i * NH * NH, H1, NN);
    k_zero<<<nblk((long)NEH * NH, 256), 256, 0, stream>>>(MBUF, (long)NEH * NH);
    k_zero<<<nblk(featTot, 256), 256, 0, stream>>>(ACC, featTot);
    k_hyper_scatter1<<<nblk((long)NNZ * 128, 256), 256, 0, stream>>>(hn, he, H1,
                                                                     MBUF);
    k_hyper_scatter2<<<nblk((long)NNZ * 128, 256), 256, 0, stream>>>(hn, he, MBUF,
                                                                     BINV, ACC);
    k_hyper_finish<<<nblk(featTot, 256), 256, 0, stream>>>(ACC, DINV,
                                                           hb + (size_t)i * NH, H0);
  }

  // ---- 4 GCN (highConv) layers; last writes f32 to d_out ----
  for (int i = 0; i < 4; ++i) {
    k_gemm512<<<ggrid, 256, 0, stream>>>(H0, GWB + (size_t)i * NH * NH, H1, NN);
    k_zero<<<nblk(featTot, 256), 256, 0, stream>>>(ACC, featTot);
    k_gcn_scatter<<<nblk((long)NE * 128, 256), 256, 0, stream>>>(es, ed, H1, GDNV,
                                                                 ACC);
    k_gcn_finish<<<nblk(featTot, 256), 256, 0, stream>>>(
        ACC, H1, GDNV, H0, (i == 3) ? outf : (float*)nullptr);
  }
}
